// CustomConv_66769561583718
// MI455X (gfx1250) — matrix operand order repaired and checked by reading, not compiled, not run on hardware
//
#include <hip/hip_runtime.h>

#define DEV_INLINE __device__ __forceinline__

// ---- feature detection for gfx1250 async-to-LDS path ----
#if defined(__has_builtin)
#  if __has_builtin(__builtin_amdgcn_global_load_async_to_lds_b128)
#    define HAS_ASYNC_LDS 1
#  endif
#  if __has_builtin(__builtin_amdgcn_s_wait_asynccnt)
#    define HAS_WAIT_ASYNC 1
#  endif
#endif
#ifndef HAS_ASYNC_LDS
#  define HAS_ASYNC_LDS 0
#endif
#ifndef HAS_WAIT_ASYNC
#  define HAS_WAIT_ASYNC 0
#endif

typedef __attribute__((ext_vector_type(16))) __bf16 v16bf;
typedef __attribute__((ext_vector_type(8)))  __bf16 v8bf;
typedef __attribute__((ext_vector_type(8)))  float  v8f;
typedef __attribute__((ext_vector_type(8)))  unsigned int vu8;
typedef __attribute__((ext_vector_type(4)))  unsigned int vu4;
typedef __attribute__((ext_vector_type(4)))  int v4i;

// ---------------- problem constants ----------------
constexpr int BATCH = 16, CIN = 128, HH = 112, WW = 112, COUT = 256;
constexpr int KTOT  = CIN * 9;                     // 1152, weight is [COUT][KTOT] row-major
constexpr int OH = 54, OW = 54, NPIX = OH * OW;    // 2916
constexpr float MU = 0.1f;

// ---------------- GEMM tiling ----------------
constexpr int MB = 128, NB = 128, KB = 32;
constexpr int KCH   = KTOT / KB;                   // 36 k-chunks
constexpr int LDSTR = KB + 8;                      // 40 ushorts = 80 B padded row
constexpr int NMT   = COUT / 16;                   // 16 m-tiles of 16 rows

DEV_INLINE unsigned rneb(float f) {                // fp32 -> bf16 bits (RNE) in [31:16]
    unsigned u = __builtin_bit_cast(unsigned, f);
    return u + 0x7fffu + ((u >> 16) & 1u);
}
DEV_INLINE float hif(unsigned r) {                 // float value of the bf16-hi part
    return __builtin_bit_cast(float, r & 0xffff0000u);
}
DEV_INLINE unsigned pkh(unsigned r0, unsigned r1) {// pack two bf16 (r0->lo, r1->hi)
    return (r0 >> 16) | (r1 & 0xffff0000u);
}

DEV_INLINE v8f wmma_bf16(v16bf a, v16bf b, v8f c) {
    return __builtin_amdgcn_wmma_f32_16x16x32_bf16(false, a, false, b, (short)0, c, false, false);
}

DEV_INLINE void async_copy_b128(const void* g, void* l) {
#if HAS_ASYNC_LDS
    __builtin_amdgcn_global_load_async_to_lds_b128(
        (__attribute__((address_space(1))) v4i*)g,
        (__attribute__((address_space(3))) v4i*)l, 0, 0);
#else
    (void)g; (void)l;
#endif
}
DEV_INLINE void wait_async_zero() {
#if HAS_WAIT_ASYNC
    __builtin_amdgcn_s_wait_asynccnt(0);
#elif HAS_ASYNC_LDS
    asm volatile("s_wait_asynccnt 0x0" ::: "memory");
#endif
}

// ---------------- kernel 0: weight -> bf16 hi/lo planes, pre-swizzled to WMMA A order --------
// layout: plane[((mt*KCH + kc)*32 + lane)*16 + e], e<8 -> K=half*8+e ; e>=8 -> K=16+half*8+(e-8)
__global__ __launch_bounds__(256) void prepw_kernel(const float* __restrict__ wgt,
                                                    unsigned short* __restrict__ whi,
                                                    unsigned short* __restrict__ wlo) {
    int idx = blockIdx.x * 256 + threadIdx.x;      // (mt*KCH + kc)*32 + lane
    if (idx >= NMT * KCH * 32) return;
    const int lane = idx & 31;
    const int mtkc = idx >> 5;
    const int kc   = mtkc % KCH;
    const int mt   = mtkc / KCH;
    const int half = lane >> 4, r = lane & 15;
    const float* src = wgt + (size_t)(mt * 16 + r) * KTOT + kc * KB;

    unsigned h[8], l[8];
#pragma unroll
    for (int e2 = 0; e2 < 8; ++e2) {
        const int e0 = 2 * e2;
        const int k0 = (e0 < 8) ? (half * 8 + e0) : (16 + half * 8 + (e0 - 8));
        const float f0 = src[k0], f1 = src[k0 + 1];
        const unsigned r0 = rneb(f0), r1 = rneb(f1);
        h[e2] = pkh(r0, r1);
        l[e2] = pkh(rneb(f0 - hif(r0)), rneb(f1 - hif(r1)));
    }
    vu4 H0 = {h[0], h[1], h[2], h[3]}, H1 = {h[4], h[5], h[6], h[7]};
    vu4 L0 = {l[0], l[1], l[2], l[3]}, L1 = {l[4], l[5], l[6], l[7]};
    *reinterpret_cast<vu4*>(whi + (size_t)idx * 16)     = H0;
    *reinterpret_cast<vu4*>(whi + (size_t)idx * 16 + 8) = H1;
    *reinterpret_cast<vu4*>(wlo + (size_t)idx * 16)     = L0;
    *reinterpret_cast<vu4*>(wlo + (size_t)idx * 16 + 8) = L1;
}

// ---------------- kernel 1: channel sum T[b,h,w] ----------------
__global__ __launch_bounds__(256) void chansum_kernel(const float* __restrict__ in,
                                                      float* __restrict__ T) {
    int idx = blockIdx.x * 256 + threadIdx.x;
    if (idx >= BATCH * HH * WW) return;
    const int b  = idx / (HH * WW);
    const int hw = idx - b * (HH * WW);
    const float* p = in + (size_t)b * CIN * HH * WW + hw;
    float s = 0.f;
#pragma unroll 8
    for (int c = 0; c < CIN; ++c) s += p[(size_t)c * HH * WW];
    T[idx] = s;
}

// ---------------- kernel 2: S[b,y,x] (fuzzy-mask conv of T, mask = MU where r|c odd) --------
__global__ __launch_bounds__(256) void fuz_kernel(const float* __restrict__ T,
                                                  float* __restrict__ S) {
    int idx = blockIdx.x * 256 + threadIdx.x;
    if (idx >= BATCH * NPIX) return;
    const int b = idx / NPIX;
    const int p = idx - b * NPIX;
    const int y = p / OW;
    const int x = p - y * OW;
    const float* t = T + (size_t)b * HH * WW + (2 * y) * WW + 2 * x;
    float s = 0.f;
#pragma unroll
    for (int r = 0; r < 5; ++r)
#pragma unroll
        for (int c = 0; c < 5; ++c)
            if ((r & 1) | (c & 1)) s += t[r * WW + c];
    S[idx] = MU * s;
}

// ---------------- kernel 3: GEMM — A frags straight from L2, B im2col double-buffered -------
__global__ __launch_bounds__(256)
void conv_gemm_wmma(const float* __restrict__ in,
                    const unsigned short* __restrict__ whi,
                    const unsigned short* __restrict__ wlo,
                    const float* __restrict__ bias,
                    const float* __restrict__ S,
                    float* __restrict__ out) {
    __shared__ __align__(16) unsigned short sBh[2][NB * LDSTR];
    __shared__ __align__(16) unsigned short sBl[2][NB * LDSTR];
    __shared__ int offTab[KTOT];
    __shared__ __align__(16) float sS[NB];
    __shared__ __align__(16) float sBias[MB];

    const int t  = threadIdx.x;
    const int b  = blockIdx.z;
    const int m0 = blockIdx.y * MB;
    const int n0 = blockIdx.x * NB;
    const int mtbase = blockIdx.y * (MB / 16);

    const int lane = t & 31;
    const int wave = t >> 5;
    const int wm   = wave >> 1;               // 0..3 : 32 output rows
    const int wn   = wave & 1;                // 0..1 : 64 output cols
    const int half = lane >> 4;
    const int lrow = lane & 15;

    // im2col offset table (element offsets into one batch image)
    for (int k = t; k < KTOT; k += 256) {
        const int ci = k / 9, r9 = k - ci * 9, kh = r9 / 3, kw = r9 - kh * 3;
        offTab[k] = ci * (HH * WW) + (2 * kh) * WW + 2 * kw;
    }

    // fire-and-forget: epilogue S / bias tiles -> LDS (drained 36 chunks later)
#if HAS_ASYNC_LDS
    if (wave == 0) {
        if (n0 + lane * 4 + 4 <= NPIX)
            async_copy_b128(&S[b * NPIX + n0 + lane * 4], &sS[lane * 4]);
        async_copy_b128(&bias[m0 + lane * 4], &sBias[lane * 4]);
    }
#else
    if (wave == 0) {
        float4 z = {0.f, 0.f, 0.f, 0.f};
        float4 sv = (n0 + lane * 4 + 4 <= NPIX)
                        ? *reinterpret_cast<const float4*>(&S[b * NPIX + n0 + lane * 4]) : z;
        *reinterpret_cast<float4*>(&sS[lane * 4]) = sv;
        *reinterpret_cast<float4*>(&sBias[lane * 4]) =
            *reinterpret_cast<const float4*>(&bias[m0 + lane * 4]);
    }
#endif
    __syncthreads();      // offTab (and fallback sS/sBias) visible

    // B staging: thread -> (pixel row nb_, k-half kg), 16 scalars/chunk
    const int nb_  = t & 127;
    const int kg   = t >> 7;
    const int pix0 = n0 + nb_;
    const bool pvalid = pix0 < NPIX;
    const int py = pvalid ? pix0 / OW : 0;
    const int px = pvalid ? (pix0 - (pix0 / OW) * OW) : 0;
    const float* inb = in + (size_t)b * CIN * HH * WW + (2 * py) * WW + 2 * px;

    float fb[16];
    auto loadB = [&](int kc) {
        const int kbase = kc * KB + kg * 16;
#pragma unroll
        for (int j = 0; j < 16; ++j)
            fb[j] = pvalid ? inb[offTab[kbase + j]] : 0.0f;
    };
    auto commitB = [&](int buf) {
        vu4 H0, H1, L0, L1;
#pragma unroll
        for (int p = 0; p < 4; ++p) {
            const unsigned r0 = rneb(fb[2 * p]),     r1 = rneb(fb[2 * p + 1]);
            const unsigned s0 = rneb(fb[2 * p + 8]), s1 = rneb(fb[2 * p + 9]);
            H0[p] = pkh(r0, r1);
            H1[p] = pkh(s0, s1);
            L0[p] = pkh(rneb(fb[2 * p] - hif(r0)),     rneb(fb[2 * p + 1] - hif(r1)));
            L1[p] = pkh(rneb(fb[2 * p + 8] - hif(s0)), rneb(fb[2 * p + 9] - hif(s1)));
        }
        const int off = nb_ * LDSTR + kg * 16;
        *reinterpret_cast<vu4*>(&sBh[buf][off])     = H0;
        *reinterpret_cast<vu4*>(&sBh[buf][off + 8]) = H1;
        *reinterpret_cast<vu4*>(&sBl[buf][off])     = L0;
        *reinterpret_cast<vu4*>(&sBl[buf][off + 8]) = L1;
    };

    v8f acc[2][4];
#pragma unroll
    for (int i = 0; i < 2; ++i)
#pragma unroll
        for (int j = 0; j < 4; ++j)
#pragma unroll
            for (int e = 0; e < 8; ++e) acc[i][j][e] = 0.f;

    loadB(0);
    for (int kc = 0; kc < KCH; ++kc) {
        const int cur = kc & 1;
        commitB(cur);                     // chunk kc: regs -> LDS bf16 hi/lo
        // A fragments straight from global (L2-resident, pre-swizzled): issue before barrier
        v16bf Ah[2], Al[2];
#pragma unroll
        for (int fm = 0; fm < 2; ++fm) {
            const int mt = mtbase + wm * 2 + fm;
            const size_t abase = ((size_t)(mt * KCH + kc) * 32 + lane) * 16;
            Ah[fm] = __builtin_bit_cast(v16bf, *reinterpret_cast<const vu8*>(whi + abase));
            Al[fm] = __builtin_bit_cast(v16bf, *reinterpret_cast<const vu8*>(wlo + abase));
        }
        if (kc + 1 < KCH) loadB(kc + 1);  // overlaps barrier + compute
        __syncthreads();                  // one barrier per chunk (double-buffered B)

#pragma unroll
        for (int fn = 0; fn < 4; ++fn) {
            const int row = wn * 64 + fn * 16 + lrow;
            const int o1  = row * LDSTR + half * 8;
            const v8bf c0 = *reinterpret_cast<const v8bf*>(&sBh[cur][o1]);
            const v8bf c1 = *reinterpret_cast<const v8bf*>(&sBh[cur][o1 + 16]);
            const v8bf d0 = *reinterpret_cast<const v8bf*>(&sBl[cur][o1]);
            const v8bf d1 = *reinterpret_cast<const v8bf*>(&sBl[cur][o1 + 16]);
            const v16bf Bh = __builtin_shufflevector(c0, c1, 0,1,2,3,4,5,6,7,8,9,10,11,12,13,14,15);
            const v16bf Bl = __builtin_shufflevector(d0, d1, 0,1,2,3,4,5,6,7,8,9,10,11,12,13,14,15);
#pragma unroll
            for (int fm = 0; fm < 2; ++fm) {
                acc[fm][fn] = wmma_bf16(Ah[fm], Bh, acc[fm][fn]);   // hi*hi
                acc[fm][fn] = wmma_bf16(Ah[fm], Bl, acc[fm][fn]);   // hi*lo
                acc[fm][fn] = wmma_bf16(Al[fm], Bh, acc[fm][fn]);   // lo*hi
            }
        }
    }

    // epilogue: drain async S/bias, add bias[o]*S and store
    wait_async_zero();
    __syncthreads();
#pragma unroll
    for (int fm = 0; fm < 2; ++fm) {
        const int olocal = wm * 32 + fm * 16 + half * 8;
        float bb[8];
#pragma unroll
        for (int r = 0; r < 8; ++r) bb[r] = sBias[olocal + r];
#pragma unroll
        for (int fn = 0; fn < 4; ++fn) {
            const int nlocal = wn * 64 + fn * 16 + lrow;
            const int pix = n0 + nlocal;
            if (pix < NPIX) {
                const float sv = sS[nlocal];
                float* op = out + ((size_t)(b * COUT + m0 + olocal)) * NPIX + pix;
#pragma unroll
                for (int r = 0; r < 8; ++r)
                    op[(size_t)r * NPIX] = acc[fm][fn][r] + bb[r] * sv;
            }
        }
    }
}

// ---------------- launcher ----------------
extern "C" void kernel_launch(void* const* d_in, const int* in_sizes, int n_in,
                              void* d_out, int out_size, void* d_ws, size_t ws_size,
                              hipStream_t stream) {
    (void)in_sizes; (void)n_in; (void)out_size; (void)ws_size;
    const float* input  = (const float*)d_in[0];
    const float* weight = (const float*)d_in[1];
    const float* bias1  = (const float*)d_in[2];
    float* out = (float*)d_out;

    float* T = (float*)d_ws;                                  // 16*112*112 f32
    float* S = T + BATCH * HH * WW;                           // 16*2916    f32
    unsigned short* whi = (unsigned short*)(S + BATCH * NPIX);// 16*36*512 bf16
    unsigned short* wlo = whi + (size_t)NMT * KCH * 32 * 16;

    prepw_kernel<<<(NMT * KCH * 32 + 255) / 256, 256, 0, stream>>>(weight, whi, wlo);
    chansum_kernel<<<(BATCH * HH * WW + 255) / 256, 256, 0, stream>>>(input, T);
    fuz_kernel<<<(BATCH * NPIX + 255) / 256, 256, 0, stream>>>(T, S);

    dim3 grid((NPIX + NB - 1) / NB, COUT / MB, BATCH);        // 23 x 2 x 16
    conv_gemm_wmma<<<grid, 256, 0, stream>>>(input, whi, wlo, bias1, S, out);
}
